// MCCVGating_79233556676728
// MI455X (gfx1250) — matrix operand order repaired and checked by reading, not compiled, not run on hardware
//
#include <hip/hip_runtime.h>
#include <hip/hip_bf16.h>
#include <math.h>

typedef __attribute__((ext_vector_type(16))) _Float16 v16h;
typedef __attribute__((ext_vector_type(8)))  _Float16 v8h;
typedef __attribute__((ext_vector_type(8)))  float    v8f;

#define BB 2
#define CH 32
#define HH 128
#define WW 240
#define DD 48
#define OC 8
#define NT (WW / 16)        // 15 M-tiles
#define DSTEP (HH * WW + 1) // sim index step per C-matrix VGPR r (d+1, w1+1)

// ---------------------------------------------------------------------------
// Kernel 0: L2-normalize along channel axis, convert to f16, repack to
// [b][h][w][c] (32 contiguous halfs per (b,h,w) column) for WMMA tile loads.
// ---------------------------------------------------------------------------
__global__ void mccv_norm_pack(const float* __restrict__ ML,
                               const float* __restrict__ MR,
                               _Float16* __restrict__ MLh,
                               _Float16* __restrict__ MRh) {
  int idx = blockIdx.x * blockDim.x + threadIdx.x;   // over B*H*W columns
  if (idx >= BB * HH * WW) return;
  int w  = idx % WW;
  int bh = idx / WW;
  int b  = bh >> 7;           // /128
  int h  = bh & 127;
  const size_t cs   = (size_t)HH * WW;               // channel stride in input
  const size_t base = ((size_t)b * CH * HH + h) * WW + w;

  float l[CH], r[CH];
  float sl = 0.f, sr = 0.f;
#pragma unroll
  for (int c = 0; c < CH; ++c) {
    l[c] = ML[base + (size_t)c * cs];
    r[c] = MR[base + (size_t)c * cs];
    sl += l[c] * l[c];
    sr += r[c] * r[c];
  }
  float il = 1.0f / fmaxf(sqrtf(sl), 1e-12f);
  float ir = 1.0f / fmaxf(sqrtf(sr), 1e-12f);

  size_t ob = (size_t)idx * CH;
#pragma unroll
  for (int c = 0; c < CH; ++c) {
    MLh[ob + c] = (_Float16)(l[c] * il);
    MRh[ob + c] = (_Float16)(r[c] * ir);
  }
}

// ---------------------------------------------------------------------------
// Kernel 1: banded correlation GEMM via V_WMMA_F32_16X16X32_F16.
// One wave32 per (b,h,M-tile). For M-tile t, N-tiles max(0,t-3)..t cover the
// band d = w1-w2 in [0,48). Band entries scattered to sim[b][d][h][w1] via a
// single per-lane base pointer + constant immediate offsets (index is affine
// in the C-VGPR index r with step DSTEP). Entries with w<d stay zero
// (workspace pre-zeroed).
// ---------------------------------------------------------------------------
__global__ void mccv_sim_wmma(const _Float16* __restrict__ MLh,
                              const _Float16* __restrict__ MRh,
                              float* __restrict__ sim) {
  const int lane = threadIdx.x & 31;
  const int wave = blockIdx.x * (blockDim.x >> 5) + (threadIdx.x >> 5);
  const int t  = wave % NT;
  const int bh = wave / NT;
  if (bh >= BB * HH) return;
  const int b = bh >> 7;
  const int h = bh & 127;

  const size_t colBase = (size_t)bh * WW * CH;       // halfs
  const int wbase = t * 16;
  const int n  = lane & 15;
  const int hi = lane >> 4;                          // 0 or 1

  // --- A tile (MLn): 16 rows (w1) x 32 (K=c), contiguous K = 16*hi + j ---
  // lane<16: M=lane,    K = {0..15}
  // lane>=16:M=lane-16, K = {16..31}
  const _Float16* aptr = MLh + colBase + (size_t)(wbase + n) * CH;
  v8h a0 = *(const v8h*)(aptr + (hi ? 16 : 0));      // 16B aligned
  v8h a1 = *(const v8h*)(aptr + (hi ? 24 : 8));
  v16h A;
#pragma unroll
  for (int j = 0; j < 8; ++j) { A[j] = a0[j]; A[8 + j] = a1[j]; }

  // Uniform part of the sim index: b*DD*HH*WW + h*WW
  const int sbase = b * DD * HH * WW + h * WW;
  const int w1_0  = wbase + (hi << 3);               // w1 for C-VGPR r=0

  int u0 = t - 3; if (u0 < 0) u0 = 0;
  for (int u = u0; u <= t; ++u) {
    // --- B tile (MRn): 32 (K=c) x 16 cols (w2). lane holds N=n, K=16*hi+j ---
    const _Float16* bptr = MRh + colBase + (size_t)(u * 16 + n) * CH + (hi ? 16 : 0);
    v16h Bm = *(const v16h*)bptr;                    // 32B contiguous, aligned

    v8f c = {};
    c = __builtin_amdgcn_wmma_f32_16x16x32_f16(
        /*neg_a=*/false, A, /*neg_b=*/false, Bm,
        /*c_mod=*/(short)0, c, /*reuse_a=*/false, /*reuse_b=*/false);

    // C/D layout: VGPR r -> (M = r + 8*hi, N = lane&15);
    // d = w1 - w2 and w1 both advance by 1 with r -> index step = DSTEP.
    const int w2 = u * 16 + n;
    const int d0 = w1_0 - w2;                        // in [-15, 48+...]
    float* p = sim + (sbase + d0 * (HH * WW) + w1_0); // may point pre-band; only
                                                      // dereferenced when valid
#pragma unroll
    for (int r = 0; r < 8; ++r) {
      if ((unsigned)(d0 + r) < (unsigned)DD) {
        p[r * DSTEP] = c[r];                         // const imm offset stores
      }
    }
  }
}

// ---------------------------------------------------------------------------
// Kernel 2: 3x3x3 SAME conv (1 -> 8 channels) + bias + sigmoid.
// One thread per (b,d,h,w); 27-tap window shared across 8 output channels.
// Weight/bias reads are wave-uniform -> scalar loads.
// ---------------------------------------------------------------------------
__global__ void mccv_conv_sigmoid(const float* __restrict__ sim,
                                  const float* __restrict__ wgt,
                                  const float* __restrict__ bias,
                                  float* __restrict__ out) {
  int idx = blockIdx.x * blockDim.x + threadIdx.x;
  if (idx >= BB * DD * HH * WW) return;
  int w = idx % WW; int tmp = idx / WW;
  int h = tmp % HH; tmp /= HH;
  int d = tmp % DD; int b = tmp / DD;

  float v[27];
  int k = 0;
#pragma unroll
  for (int kd = -1; kd <= 1; ++kd)
#pragma unroll
    for (int kh = -1; kh <= 1; ++kh)
#pragma unroll
      for (int kw = -1; kw <= 1; ++kw, ++k) {
        int dd = d + kd, hh = h + kh, ww = w + kw;
        bool ok = (dd >= 0) & (dd < DD) & (hh >= 0) & (hh < HH) &
                  (ww >= 0) & (ww < WW);
        v[k] = ok ? sim[(((size_t)b * DD + dd) * HH + hh) * WW + ww] : 0.0f;
      }

#pragma unroll
  for (int o = 0; o < OC; ++o) {
    float acc = bias[o];
#pragma unroll
    for (int k2 = 0; k2 < 27; ++k2) acc = fmaf(wgt[o * 27 + k2], v[k2], acc);
    out[((((size_t)b * OC + o) * DD + d) * HH + h) * WW + w] =
        1.0f / (1.0f + __expf(-acc));
  }
}

// ---------------------------------------------------------------------------
extern "C" void kernel_launch(void* const* d_in, const int* in_sizes, int n_in,
                              void* d_out, int out_size, void* d_ws, size_t ws_size,
                              hipStream_t stream) {
  const float* ML     = (const float*)d_in[0];
  const float* MR     = (const float*)d_in[1];
  const float* conv_w = (const float*)d_in[2];
  const float* conv_b = (const float*)d_in[3];
  // d_in[4] = D (scalar, 48) — baked into the kernels.

  const size_t nCols   = (size_t)BB * HH * WW;       // 61440
  _Float16* MLh = (_Float16*)d_ws;                   // nCols*32 halfs ≈ 3.93 MB
  _Float16* MRh = MLh + nCols * CH;
  float*    sim = (float*)(MRh + nCols * CH);        // B*D*H*W f32 ≈ 11.25 MB
  const size_t simElems = (size_t)BB * DD * HH * WW;

  // Zero the band complement (w < d stays 0); graph-capture safe.
  hipMemsetAsync(sim, 0, simElems * sizeof(float), stream);

  mccv_norm_pack<<<(int)((nCols + 255) / 256), 256, 0, stream>>>(ML, MR, MLh, MRh);

  // 2*128*15 = 3840 waves, 4 waves per 128-thread block -> 960 blocks.
  mccv_sim_wmma<<<960, 128, 0, stream>>>(MLh, MRh, sim);

  mccv_conv_sigmoid<<<(int)((simElems + 255) / 256), 256, 0, stream>>>(
      sim, conv_w, conv_b, (float*)d_out);
}
